// MOERTDETR_2147483648648
// MI455X (gfx1250) — matrix operand order repaired and checked by reading, not compiled, not run on hardware
//
#include <hip/hip_runtime.h>
#include <hip/hip_bf16.h>

// MoE fused kernel for MI455X (gfx1250), wave32 + WMMA bf16.
// Tokens N = 8*4096 = 32768, D = H = 256, E = 6, ROUTER_H = 128, TOP_K = 2.

typedef __attribute__((ext_vector_type(16))) __bf16 v16bf;
typedef __attribute__((ext_vector_type(8)))  float  v8f;

#define NTOK   32768
#define BM     64          // tokens per block
#define DDIM   256
#define HDIM   256
#define NEXP   6
#define RHID   128
#define LDW    264         // padded LDS row stride (halves): 4-bank rotation per row

union frag16 { uint4 q[2]; v16bf v; };

__device__ __forceinline__ unsigned short f2bf(float x) {
  unsigned int u = __builtin_bit_cast(unsigned int, x);
  unsigned int r = u + 0x7FFFu + ((u >> 16) & 1u);   // round-to-nearest-even
  return (unsigned short)(r >> 16);
}
__device__ __forceinline__ float bf2f(unsigned short h) {
  unsigned int u = ((unsigned int)h) << 16;
  return __builtin_bit_cast(float, u);
}
__device__ __forceinline__ v8f v8f_zero() {
  v8f z;
#pragma unroll
  for (int i = 0; i < 8; ++i) z[i] = 0.f;
  return z;
}

// A fragment (16x32 bf16): lanes 0-15 row M=lane, K=k0..k0+15;
// lanes 16-31 row M=lane-16, K=k0+16..k0+31. Two ds_load_b128 per fragment.
__device__ __forceinline__ v16bf load_a_lds(const unsigned short* A, int m0, int k0, int lane) {
  frag16 f;
  const int row = m0 + (lane & 15);
  const int kb  = k0 + ((lane & 16) ? 16 : 0);
  const unsigned short* p = A + row * LDW + kb;
  f.q[0] = *reinterpret_cast<const uint4*>(p);
  f.q[1] = *reinterpret_cast<const uint4*>(p + 8);
  return f.v;
}

// B fragment (32x16 bf16): lane n = n0+(lane&15); lanes 0-15 hold K=k0..k0+15,
// lanes 16-31 hold K=k0+16..k0+31 (2 halves per VGPR, K-contiguous).
// Weights pre-transposed to [N][K=256] bf16 in ws -> 32 contiguous bytes per lane.
__device__ __forceinline__ v16bf load_b_glb(const unsigned short* __restrict__ Wt,
                                            int n0, int k0, int lane) {
  frag16 f;
  const int n  = n0 + (lane & 15);
  const int kb = k0 + ((lane & 16) ? 16 : 0);
  const uint4* p = reinterpret_cast<const uint4*>(Wt + n * 256 + kb);
  f.q[0] = p[0];
  f.q[1] = p[1];
  return f.v;
}

// Core GEMM: acc[NTPW][4] += A[BM x 256] * Wt^T for this wave's NTPW n-tiles.
// A fragments are loaded from LDS ONCE per (kt, mt) and reused across all n-tiles
// (halves A-side LDS traffic per WMMA vs. per-n-tile passes).
template <int NTPW>
__device__ __forceinline__ void gemm_core(const unsigned short* __restrict__ A,
                                          const unsigned short* __restrict__ Wt,
                                          v8f acc[NTPW][4], int wave, int lane) {
  const int nbase = wave * NTPW * 16;
#pragma unroll
  for (int kt = 0; kt < 8; ++kt) {
    v16bf b[NTPW];
#pragma unroll
    for (int nt = 0; nt < NTPW; ++nt)
      b[nt] = load_b_glb(Wt, nbase + nt * 16, kt * 32, lane);
#pragma unroll
    for (int mt = 0; mt < 4; ++mt) {
      const v16bf a = load_a_lds(A, mt * 16, kt * 32, lane);
#pragma unroll
      for (int nt = 0; nt < NTPW; ++nt)
        acc[nt][mt] = __builtin_amdgcn_wmma_f32_16x16x32_bf16(
            false, a, false, b[nt], (short)0, acc[nt][mt], false, false);
    }
  }
}

// GEMM + bias + ReLU, bf16 result stored to LDS.
template <int NTPW>
__device__ __forceinline__ void gemm_relu_store(const unsigned short* __restrict__ A,
                                                const unsigned short* __restrict__ Wt,
                                                const float* __restrict__ bias,
                                                unsigned short* __restrict__ Dst,
                                                int wave, int lane) {
  v8f acc[NTPW][4];
#pragma unroll
  for (int nt = 0; nt < NTPW; ++nt)
#pragma unroll
    for (int mt = 0; mt < 4; ++mt) acc[nt][mt] = v8f_zero();

  gemm_core<NTPW>(A, Wt, acc, wave, lane);

  // C/D layout: lane holds (M = r + (lane>=16 ? 8:0), N = lane&15), r = 0..7.
  const int rbase = (lane & 16) ? 8 : 0;
#pragma unroll
  for (int nt = 0; nt < NTPW; ++nt) {
    const int col  = (wave * NTPW + nt) * 16 + (lane & 15);
    const float bc = bias[col];
#pragma unroll
    for (int mt = 0; mt < 4; ++mt)
#pragma unroll
      for (int r = 0; r < 8; ++r) {
        float v = acc[nt][mt][r] + bc;
        v = v > 0.f ? v : 0.f;
        Dst[(mt * 16 + rbase + r) * LDW + col] = f2bf(v);
      }
  }
}

// Final GEMM: facc += gate[token][e] * (A * W3t^T + b3). 2 n-tiles per wave.
__device__ __forceinline__ void gemm_gate_acc(const unsigned short* __restrict__ A,
                                              const unsigned short* __restrict__ Wt,
                                              const float* __restrict__ bias,
                                              const float* __restrict__ gates, int e,
                                              v8f facc[2][4], int wave, int lane) {
  v8f acc[2][4];
#pragma unroll
  for (int nt = 0; nt < 2; ++nt)
#pragma unroll
    for (int mt = 0; mt < 4; ++mt) acc[nt][mt] = v8f_zero();

  gemm_core<2>(A, Wt, acc, wave, lane);

  const int rbase = (lane & 16) ? 8 : 0;
#pragma unroll
  for (int nt = 0; nt < 2; ++nt) {
    const float bc = bias[(wave * 2 + nt) * 16 + (lane & 15)];
#pragma unroll
    for (int mt = 0; mt < 4; ++mt)
#pragma unroll
      for (int r = 0; r < 8; ++r) {
        const int row = mt * 16 + rbase + r;
        facc[nt][mt][r] += gates[row * 8 + e] * (acc[nt][mt][r] + bc);
      }
  }
}

// ---- weight prep: fp32 -> bf16 with transpose to [N][K] layout ----
// ws halves layout: [0, 32768)              router_w1^T [128][256]
//                   [32768 + e*65536)       expert_w1^T [e][256][256]
//                   [+6*65536 ...)          expert_w2^T
//                   [+12*65536 ...)         expert_w3^T
#define WS_TOTAL (32768 + 18 * 65536)

__global__ __launch_bounds__(256) void prep_weights(const float* __restrict__ rw1,
                                                    const float* __restrict__ ew1,
                                                    const float* __restrict__ ew2,
                                                    const float* __restrict__ ew3,
                                                    unsigned short* __restrict__ wt) {
  const int id = blockIdx.x * 256 + threadIdx.x;
  if (id >= WS_TOTAL) return;
  float v;
  if (id < 32768) {
    const int n = id >> 8, k = id & 255;      // router_w1 [256][128] -> [128][256]
    v = rw1[k * RHID + n];
  } else {
    const int t = id - 32768;
    const int mat = t >> 16;                  // 0..17
    const int n = (t & 65535) >> 8, k = t & 255;
    const float* src = (mat < 6) ? ew1 : ((mat < 12) ? ew2 : ew3);
    const int e = mat % 6;
    v = src[(e * 256 + k) * 256 + n];         // all expert mats are [256][256]
  }
  wt[id] = f2bf(v);
}

// ---- fused MoE kernel: router + top-2 gates + 6 experts, 64 tokens/block ----
__global__ __launch_bounds__(256) void moe_fused_kernel(
    const float* __restrict__ feat, const float* __restrict__ rb1,
    const float* __restrict__ rw2, const float* __restrict__ rb2,
    const float* __restrict__ eb1, const float* __restrict__ eb2,
    const float* __restrict__ eb3, const unsigned short* __restrict__ wt,
    float* __restrict__ out) {
  __shared__ unsigned short Xs[BM * LDW];    // features, bf16
  __shared__ unsigned short H1s[BM * LDW];   // router hidden / expert h1
  __shared__ unsigned short H2s[BM * LDW];   // expert h2
  __shared__ float gates[BM * 8];            // dense gate per token (0 if not top-2)

  const int tid  = threadIdx.x;
  const int lane = tid & 31;
  const int wave = tid >> 5;
  const int t0   = blockIdx.x * BM;

  // Stage features fp32 -> LDS bf16 (64 x 256, padded rows).
  {
    const float4* src = reinterpret_cast<const float4*>(feat + (long)t0 * DDIM);
#pragma unroll
    for (int i = 0; i < 16; ++i) {
      const int f  = tid + i * 256;          // 0..4095 float4 slots
      const int m  = f >> 6;
      const int d0 = (f & 63) * 4;
      const float4 v = src[f];
      unsigned short* dst = Xs + m * LDW + d0;
      dst[0] = f2bf(v.x); dst[1] = f2bf(v.y); dst[2] = f2bf(v.z); dst[3] = f2bf(v.w);
    }
  }
  __syncthreads();

  // Router hidden layer: RH = relu(X @ Wr1 + br1), N=128 -> 1 n-tile per wave.
  gemm_relu_store<1>(Xs, wt, rb1, H1s, wave, lane);
  __syncthreads();

  // Router logits (128 -> 6), softmax, top-2, renormalizing softmax -> dense gates.
  if (tid < BM) {
    float lg[NEXP];
#pragma unroll
    for (int e = 0; e < NEXP; ++e) lg[e] = rb2[e];
    for (int k = 0; k < RHID; ++k) {
      const float rv = bf2f(H1s[tid * LDW + k]);
#pragma unroll
      for (int e = 0; e < NEXP; ++e) lg[e] += rv * rw2[k * NEXP + e];
    }
    float mx = lg[0];
#pragma unroll
    for (int e = 1; e < NEXP; ++e) mx = fmaxf(mx, lg[e]);
    float p[NEXP], s = 0.f;
#pragma unroll
    for (int e = 0; e < NEXP; ++e) { p[e] = __expf(lg[e] - mx); s += p[e]; }
    const float inv_s = 1.f / s;
#pragma unroll
    for (int e = 0; e < NEXP; ++e) p[e] *= inv_s;
    int i1 = 0;
#pragma unroll
    for (int e = 1; e < NEXP; ++e) if (p[e] > p[i1]) i1 = e;
    int i2 = (i1 == 0) ? 1 : 0;
#pragma unroll
    for (int e = 0; e < NEXP; ++e) if (e != i1 && p[e] > p[i2]) i2 = e;
    // reference: routing_weights = softmax over the two top softmax probabilities
    const float e1 = __expf(p[i1]), e2 = __expf(p[i2]);
    const float inv = 1.f / (e1 + e2);
#pragma unroll
    for (int e = 0; e < NEXP; ++e) gates[tid * 8 + e] = 0.f;
    gates[tid * 8 + i1] = e1 * inv;
    gates[tid * 8 + i2] = e2 * inv;
  }
  __syncthreads();

  v8f facc[2][4];
#pragma unroll
  for (int nt = 0; nt < 2; ++nt)
#pragma unroll
    for (int mt = 0; mt < 4; ++mt) facc[nt][mt] = v8f_zero();

  const unsigned short* w1t = wt + 32768;
  const unsigned short* w2t = wt + 32768 + 6 * 65536;
  const unsigned short* w3t = wt + 32768 + 12 * 65536;

  for (int e = 0; e < NEXP; ++e) {
    gemm_relu_store<2>(Xs,  w1t + e * 65536, eb1 + e * HDIM, H1s, wave, lane);
    __syncthreads();
    gemm_relu_store<2>(H1s, w2t + e * 65536, eb2 + e * HDIM, H2s, wave, lane);
    __syncthreads();
    gemm_gate_acc(H2s, w3t + e * 65536, eb3 + e * DDIM, gates, e, facc, wave, lane);
    __syncthreads();
  }

  // Write combined output (fp32).
  const int rbase = (lane & 16) ? 8 : 0;
#pragma unroll
  for (int nt = 0; nt < 2; ++nt) {
    const int col = (wave * 2 + nt) * 16 + (lane & 15);
#pragma unroll
    for (int mt = 0; mt < 4; ++mt)
#pragma unroll
      for (int r = 0; r < 8; ++r) {
        const int row = mt * 16 + rbase + r;
        out[(long)(t0 + row) * DDIM + col] = facc[nt][mt][r];
      }
  }
}

extern "C" void kernel_launch(void* const* d_in, const int* in_sizes, int n_in,
                              void* d_out, int out_size, void* d_ws, size_t ws_size,
                              hipStream_t stream) {
  (void)in_sizes; (void)n_in; (void)out_size; (void)ws_size;
  const float* feat = (const float*)d_in[0];
  const float* rw1  = (const float*)d_in[1];
  const float* rb1  = (const float*)d_in[2];
  const float* rw2  = (const float*)d_in[3];
  const float* rb2  = (const float*)d_in[4];
  const float* ew1  = (const float*)d_in[5];
  const float* eb1  = (const float*)d_in[6];
  const float* ew2  = (const float*)d_in[7];
  const float* eb2  = (const float*)d_in[8];
  const float* ew3  = (const float*)d_in[9];
  const float* eb3  = (const float*)d_in[10];
  unsigned short* wt = (unsigned short*)d_ws;
  float* out = (float*)d_out;

  prep_weights<<<(WS_TOTAL + 255) / 256, 256, 0, stream>>>(rw1, ew1, ew2, ew3, wt);
  moe_fused_kernel<<<NTOK / BM, 256, 0, stream>>>(feat, rb1, rw2, rb2, eb1, eb2, eb3,
                                                  wt, out);
}
